// EVEMixtralSparseBlock_69123203662111
// MI455X (gfx1250) — matrix operand order repaired and checked by reading, not compiled, not run on hardware
//
#include <hip/hip_runtime.h>
#include <hip/hip_bf16.h>

typedef __bf16 bf16;
typedef __attribute__((ext_vector_type(16))) __bf16 v16bf;
typedef __attribute__((ext_vector_type(8)))  __bf16 v8bf;
typedef __attribute__((ext_vector_type(8)))  float   v8f;
typedef __attribute__((ext_vector_type(4)))  int     v4i;

#define N_TOK 4096
#define H_DIM 4096
#define F_DIM 14336
#define N_EXP 8
#define R_LORA 8
#define LORA_SCALE 2.0f

// LDS row stride (elements): 40 (80B = 20 dwords; 20*r mod 64 distinct for r=0..15
// -> conflict-free b128 fragment reads across each 16-lane half)
#define LDS_STRIDE 40

// --------------------------------------------------------------------------
// gfx1250 async global->LDS path (probe-confirmed):
//   void __builtin_amdgcn_global_load_async_to_lds_b128(v4i as(1)*, v4i as(3)*,
//                                                       imm offset, imm cpol)
// --------------------------------------------------------------------------
#if defined(__has_builtin)
#if __has_builtin(__builtin_amdgcn_global_load_async_to_lds_b128)
#define USE_ASYNC_LDS 1
#endif
#if __has_builtin(__builtin_amdgcn_s_wait_asynccnt)
#define HAVE_WAIT_ASYNC_BUILTIN 1
#endif
#endif
#ifndef USE_ASYNC_LDS
#define USE_ASYNC_LDS 0
#endif

typedef __attribute__((address_space(1))) v4i g_v4i;
typedef __attribute__((address_space(3))) v4i l_v4i;

__device__ __forceinline__ void async_cp16(const bf16* gsrc, bf16* ldst) {
#if USE_ASYNC_LDS
  // int->pointer casts sidestep generic->AS addrspacecast restrictions;
  // generic LDS pointer carries the LDS offset in its low 32 bits.
  g_v4i* gp = (g_v4i*)(unsigned long long)gsrc;
  l_v4i* lp = (l_v4i*)(unsigned int)(unsigned long long)ldst;
  __builtin_amdgcn_global_load_async_to_lds_b128(gp, lp, 0, 0);
#else
  *(uint4*)ldst = *(const uint4*)gsrc;
#endif
}

__device__ __forceinline__ void wait_async_all() {
#if USE_ASYNC_LDS
#if defined(HAVE_WAIT_ASYNC_BUILTIN)
  __builtin_amdgcn_s_wait_asynccnt(0);
#else
  asm volatile("s_wait_asynccnt 0x0" ::: "memory");
#endif
#endif
}

// ---------------------------------------------------------------------------
// Convert f32 -> bf16 (flat, 4 elems/thread)
// ---------------------------------------------------------------------------
__global__ __launch_bounds__(256) void cvt_bf16_kernel(const float* __restrict__ in,
                                                       bf16* __restrict__ out,
                                                       unsigned long long n) {
  unsigned long long i = ((unsigned long long)blockIdx.x * 256ull + threadIdx.x) * 4ull;
  if (i + 3 < n) {
    float4 v = *(const float4*)(in + i);
    out[i + 0] = (bf16)v.x;
    out[i + 1] = (bf16)v.y;
    out[i + 2] = (bf16)v.z;
    out[i + 3] = (bf16)v.w;
  }
}

// ---------------------------------------------------------------------------
// Tiled transpose + convert: in [R,C] f32 row-major -> out [C,R] bf16 row-major
// ---------------------------------------------------------------------------
__global__ __launch_bounds__(256) void transpose_cvt_kernel(const float* __restrict__ in,
                                                            bf16* __restrict__ out,
                                                            int R, int C) {
  __shared__ float tile[32][33];
  const int tx = threadIdx.x & 31;
  const int ty = threadIdx.x >> 5;  // 0..7
  const int c0 = blockIdx.x * 32;
  const int r0 = blockIdx.y * 32;
#pragma unroll
  for (int i = 0; i < 4; ++i)
    tile[ty + i * 8][tx] = in[(size_t)(r0 + ty + i * 8) * C + c0 + tx];
  __syncthreads();
#pragma unroll
  for (int i = 0; i < 4; ++i)
    out[(size_t)(c0 + ty + i * 8) * R + r0 + tx] = (bf16)tile[tx][ty + i * 8];
}

// ---------------------------------------------------------------------------
// Router: wave per token. logits -> d_out tail; top-2 weights/indices -> ws
// ---------------------------------------------------------------------------
__global__ __launch_bounds__(256) void router_kernel(const float* __restrict__ x,
                                                     const float* __restrict__ rweight,
                                                     float* __restrict__ logits,
                                                     float* __restrict__ topw,
                                                     int* __restrict__ topi) {
  const int wid = threadIdx.x >> 5;
  const int lane = threadIdx.x & 31;
  const int n = blockIdx.x * 8 + wid;

  const float4* xr = (const float4*)(x + (size_t)n * H_DIM);
  float acc[N_EXP];
#pragma unroll
  for (int e = 0; e < N_EXP; ++e) acc[e] = 0.0f;

  for (int i = lane; i < H_DIM / 4; i += 32) {
    float4 xv = xr[i];
#pragma unroll
    for (int e = 0; e < N_EXP; ++e) {
      float4 wv = ((const float4*)(rweight + (size_t)e * H_DIM))[i];
      acc[e] += xv.x * wv.x + xv.y * wv.y + xv.z * wv.z + xv.w * wv.w;
    }
  }
#pragma unroll
  for (int e = 0; e < N_EXP; ++e)
#pragma unroll
    for (int off = 16; off > 0; off >>= 1)
      acc[e] += __shfl_xor(acc[e], off, 32);

  if (lane == 0) {
    int e0 = 0;
    float b0 = acc[0];
    for (int e = 1; e < N_EXP; ++e)
      if (acc[e] > b0) { b0 = acc[e]; e0 = e; }
    int e1 = (e0 == 0) ? 1 : 0;
    float b1 = acc[e1];
    for (int e = 0; e < N_EXP; ++e)
      if (e != e0 && acc[e] > b1) { b1 = acc[e]; e1 = e; }
    // renormalized top-2 softmax weight: exp(b0)/(exp(b0)+exp(b1))
    float w0 = 1.0f / (1.0f + __expf(b1 - b0));
    topw[n * 2 + 0] = w0;
    topw[n * 2 + 1] = 1.0f - w0;
    topi[n * 2 + 0] = e0;
    topi[n * 2 + 1] = e1;
#pragma unroll
    for (int e = 0; e < N_EXP; ++e) logits[(size_t)n * N_EXP + e] = acc[e];
  }
}

// ---------------------------------------------------------------------------
// WMMA bf16 GEMM. A:[M,K] row-major bf16. B0t/B1t: [N,K] row-major (B^T).
// NB==2: fused dual-GEMM + SwiGLU epilogue -> bf16 out.
// NB==1: single GEMM -> f32 out.
// Block tile 64(M) x 128(N), BK=32, 8 waves each 32x32.
// blockIdx.x sweeps M (fastest) so concurrent blocks share one 4MB B-slab and
// the whole A operand cycles through the 192MB L2 -> ~single HBM read of each.
// Double-buffered, bank-conflict-free LDS; async global->LDS (ASYNCcnt).
// ---------------------------------------------------------------------------
__device__ __forceinline__ v16bf ld_frag16(const bf16* p0, const bf16* p1) {
  v8bf lo = *(const v8bf*)p0;
  v8bf hi = *(const v8bf*)p1;
  return __builtin_shufflevector(lo, hi, 0, 1, 2, 3, 4, 5, 6, 7, 8, 9, 10, 11, 12, 13, 14, 15);
}

template <int NB>
__global__ __launch_bounds__(256) void gemm_bf16_kernel(const bf16* __restrict__ A,
                                                        const bf16* __restrict__ B0t,
                                                        const bf16* __restrict__ B1t,
                                                        bf16* __restrict__ outBf,
                                                        float* __restrict__ outF,
                                                        int M, int N, int K) {
  __shared__ __align__(16) bf16 lsA[2][64 * LDS_STRIDE];
  __shared__ __align__(16) bf16 lsB0[2][128 * LDS_STRIDE];
  __shared__ __align__(16) bf16 lsB1[2][(NB == 2) ? 128 * LDS_STRIDE : 8];

  const int tid  = threadIdx.x;
  const int lane = tid & 31;
  const int wid  = tid >> 5;
  const int wr   = wid >> 2;          // 0..1  (M direction, 32 rows each)
  const int wc   = wid & 3;           // 0..3  (N direction, 32 cols each)
  const int hsel = (lane >> 4) & 1;   // lane half select
  const int l16  = lane & 15;

  const int m0 = blockIdx.x * 64;     // M fastest-varying for L2 B-slab reuse
  const int n0 = blockIdx.y * 128;

  // global->LDS staging assignments
  const int rowA = tid >> 2;            // 0..63
  const int colA = (tid & 3) * 8;       // 0,8,16,24
  const int rowB = tid >> 1;            // 0..127
  const int colB = (tid & 1) * 16;      // 0,16

  const size_t aBase = (size_t)(m0 + rowA) * K + colA;
  const size_t bBase = (size_t)(n0 + rowB) * K + colB;

  v8f acc0[2][2] = {};
  v8f acc1[2][2] = {};

  auto stage = [&](int buf, int k0) {
    async_cp16(&A[aBase + k0], &lsA[buf][rowA * LDS_STRIDE + colA]);
    async_cp16(&B0t[bBase + k0], &lsB0[buf][rowB * LDS_STRIDE + colB]);
    async_cp16(&B0t[bBase + k0 + 8], &lsB0[buf][rowB * LDS_STRIDE + colB + 8]);
    if (NB == 2) {
      async_cp16(&B1t[bBase + k0], &lsB1[buf][rowB * LDS_STRIDE + colB]);
      async_cp16(&B1t[bBase + k0 + 8], &lsB1[buf][rowB * LDS_STRIDE + colB + 8]);
    }
  };

  stage(0, 0);
  int buf = 0;
  for (int k0 = 0; k0 < K; k0 += 32) {
    wait_async_all();     // my async loads into `buf` are done
    __syncthreads();      // everyone's are; everyone finished reading buf^1

    if (k0 + 32 < K) stage(buf ^ 1, k0 + 32);  // overlap next tile with compute

    v16bf af[2];
    v16bf b0f[2];
    v16bf b1f[2] = {};
#pragma unroll
    for (int mi = 0; mi < 2; ++mi) {
      // A frag: lane<16 -> row m_off+l16, K0..15; lane>=16 -> K16..31
      const bf16* base = &lsA[buf][(wr * 32 + mi * 16 + l16) * LDS_STRIDE];
      af[mi] = ld_frag16(base + hsel * 16, base + hsel * 16 + 8);
    }
#pragma unroll
    for (int ni = 0; ni < 2; ++ni) {
      // B frag: lane<16 -> col n_off+l16, K0..15; lane>=16 -> K16..31 (contiguous in B^T)
      const bf16* base0 = &lsB0[buf][(wc * 32 + ni * 16 + l16) * LDS_STRIDE + hsel * 16];
      b0f[ni] = ld_frag16(base0, base0 + 8);
      if (NB == 2) {
        const bf16* base1 = &lsB1[buf][(wc * 32 + ni * 16 + l16) * LDS_STRIDE + hsel * 16];
        b1f[ni] = ld_frag16(base1, base1 + 8);
      }
    }
#pragma unroll
    for (int mi = 0; mi < 2; ++mi)
#pragma unroll
      for (int ni = 0; ni < 2; ++ni) {
        acc0[mi][ni] = __builtin_amdgcn_wmma_f32_16x16x32_bf16(
            false, af[mi], false, b0f[ni], (short)0, acc0[mi][ni], false, false);
        if (NB == 2)
          acc1[mi][ni] = __builtin_amdgcn_wmma_f32_16x16x32_bf16(
              false, af[mi], false, b1f[ni], (short)0, acc1[mi][ni], false, false);
      }
    buf ^= 1;
  }

  // Epilogue. C/D layout: VGPR r: lane<16 -> M=r, N=lane; lane>=16 -> M=r+8, N=lane-16
#pragma unroll
  for (int mi = 0; mi < 2; ++mi)
#pragma unroll
    for (int ni = 0; ni < 2; ++ni) {
      const int gr = m0 + wr * 32 + mi * 16 + hsel * 8;
      const int gc = n0 + wc * 32 + ni * 16 + l16;
      if (NB == 2) {
#pragma unroll
        for (int r = 0; r < 8; ++r) {
          float h1 = acc0[mi][ni][r];
          float h3 = acc1[mi][ni][r];
          // silu(h1)*h3 with hardware reciprocal (v_rcp_f32)
          float g = h1 * __builtin_amdgcn_rcpf(1.0f + __expf(-h1)) * h3;
          outBf[(size_t)(gr + r) * N + gc] = (bf16)g;
        }
      } else {
#pragma unroll
        for (int r = 0; r < 8; ++r)
          outF[(size_t)(gr + r) * N + gc] = acc0[mi][ni][r];
      }
    }
}

// ---------------------------------------------------------------------------
// LoRA for the 2 selected experts, wave per token; out[n,:] += sum contributions
// ---------------------------------------------------------------------------
__global__ __launch_bounds__(256) void lora_kernel(const float* __restrict__ x,
                                                   const float* __restrict__ lA,
                                                   const float* __restrict__ lB,
                                                   const float* __restrict__ topw,
                                                   const int* __restrict__ topi,
                                                   float* __restrict__ out) {
  const int wid = threadIdx.x >> 5;
  const int lane = threadIdx.x & 31;
  const int n = blockIdx.x * 8 + wid;

  const int e0 = topi[n * 2 + 0];
  const int e1 = topi[n * 2 + 1];
  const float w0 = topw[n * 2 + 0];
  const float w1 = topw[n * 2 + 1];

  const float* xr = x + (size_t)n * H_DIM;
  float a0[R_LORA], a1[R_LORA];
#pragma unroll
  for (int r = 0; r < R_LORA; ++r) { a0[r] = 0.0f; a1[r] = 0.0f; }

  // a[e] = x . lora_A[e]  (lora_A: [E,H,R], h-major, R contiguous)
  for (int h = lane; h < H_DIM; h += 32) {
    float xv = xr[h];
    const float4* A0 = (const float4*)(lA + ((size_t)e0 * H_DIM + h) * R_LORA);
    const float4* A1 = (const float4*)(lA + ((size_t)e1 * H_DIM + h) * R_LORA);
    float4 p = A0[0], q = A0[1];
    a0[0] += xv * p.x; a0[1] += xv * p.y; a0[2] += xv * p.z; a0[3] += xv * p.w;
    a0[4] += xv * q.x; a0[5] += xv * q.y; a0[6] += xv * q.z; a0[7] += xv * q.w;
    p = A1[0]; q = A1[1];
    a1[0] += xv * p.x; a1[1] += xv * p.y; a1[2] += xv * p.z; a1[3] += xv * p.w;
    a1[4] += xv * q.x; a1[5] += xv * q.y; a1[6] += xv * q.z; a1[7] += xv * q.w;
  }
#pragma unroll
  for (int r = 0; r < R_LORA; ++r) {
#pragma unroll
    for (int off = 16; off > 0; off >>= 1) {
      a0[r] += __shfl_xor(a0[r], off, 32);
      a1[r] += __shfl_xor(a1[r], off, 32);
    }
  }

  float c0[R_LORA], c1[R_LORA];
#pragma unroll
  for (int r = 0; r < R_LORA; ++r) {
    c0[r] = LORA_SCALE * w0 * a0[r];
    c1[r] = LORA_SCALE * w1 * a1[r];
  }

  // out[n,h] += sum_r c0[r]*B[e0,r,h] + c1[r]*B[e1,r,h]   (lora_B: [E,R,H])
  float* orow = out + (size_t)n * H_DIM;
  for (int h = lane * 4; h < H_DIM; h += 128) {
    float4 acc = *(float4*)(orow + h);
#pragma unroll
    for (int r = 0; r < R_LORA; ++r) {
      float4 b0 = *(const float4*)(lB + ((size_t)e0 * R_LORA + r) * H_DIM + h);
      acc.x += c0[r] * b0.x; acc.y += c0[r] * b0.y;
      acc.z += c0[r] * b0.z; acc.w += c0[r] * b0.w;
      float4 b1 = *(const float4*)(lB + ((size_t)e1 * R_LORA + r) * H_DIM + h);
      acc.x += c1[r] * b1.x; acc.y += c1[r] * b1.y;
      acc.z += c1[r] * b1.z; acc.w += c1[r] * b1.w;
    }
    *(float4*)(orow + h) = acc;
  }
}

// ---------------------------------------------------------------------------
// Launcher
// ---------------------------------------------------------------------------
extern "C" void kernel_launch(void* const* d_in, const int* in_sizes, int n_in,
                              void* d_out, int out_size, void* d_ws, size_t ws_size,
                              hipStream_t stream) {
  (void)in_sizes; (void)n_in; (void)out_size; (void)ws_size;

  const float* x        = (const float*)d_in[0];  // [N_TOK, H]
  const float* router_w = (const float*)d_in[1];  // [E, H]
  const float* w1       = (const float*)d_in[2];  // [H, F]
  const float* w2       = (const float*)d_in[3];  // [F, H]
  const float* w3       = (const float*)d_in[4];  // [H, F]
  const float* lA       = (const float*)d_in[5];  // [E, H, R]
  const float* lB       = (const float*)d_in[6];  // [E, R, H]

  float* out    = (float*)d_out;                          // [N_TOK, H]
  float* logits = out + (size_t)N_TOK * H_DIM;            // [N_TOK, E]

  // workspace layout (bytes)
  char* ws = (char*)d_ws;
  const size_t sz_xb  = (size_t)N_TOK * H_DIM * 2;  // 33.5 MB
  const size_t sz_w   = (size_t)H_DIM * F_DIM * 2;  // 117.4 MB each
  bf16* xb   = (bf16*)(ws);
  bf16* w1t  = (bf16*)(ws + sz_xb);                 // [F, H]  (w1^T)
  bf16* w3t  = (bf16*)(ws + sz_xb + sz_w);          // [F, H]  (w3^T)
  bf16* w2t  = (bf16*)(ws + sz_xb + 2 * sz_w);      // [H, F]  (w2^T)
  bf16* g    = (bf16*)(ws + sz_xb + 3 * sz_w);      // [N_TOK, F]
  float* topw = (float*)(ws + sz_xb + 4 * sz_w);
  int*   topi = (int*)(ws + sz_xb + 4 * sz_w + (size_t)N_TOK * 2 * 4);

  // 1) precision conversion (+ weight transposes so WMMA B-frags are contiguous-K)
  {
    unsigned long long n = (unsigned long long)N_TOK * H_DIM;
    cvt_bf16_kernel<<<(unsigned)(n / 4 / 256), 256, 0, stream>>>(x, xb, n);
  }
  {
    dim3 tg(F_DIM / 32, H_DIM / 32);
    transpose_cvt_kernel<<<tg, 256, 0, stream>>>(w1, w1t, H_DIM, F_DIM);
    transpose_cvt_kernel<<<tg, 256, 0, stream>>>(w3, w3t, H_DIM, F_DIM);
  }
  {
    dim3 tg(H_DIM / 32, F_DIM / 32);
    transpose_cvt_kernel<<<tg, 256, 0, stream>>>(w2, w2t, F_DIM, H_DIM);
  }

  // 2) router (logits to d_out tail; top-2 to ws)
  router_kernel<<<N_TOK / 8, 256, 0, stream>>>(x, router_w, logits, topw, topi);

  // 3) fused dual GEMM + SwiGLU: g = silu(x@w1) * (x@w3)   [N_TOK, F] bf16
  {
    dim3 grid(N_TOK / 64, F_DIM / 128);   // x: M (fastest), y: N
    gemm_bf16_kernel<2><<<grid, 256, 0, stream>>>(xb, w1t, w3t, g, nullptr,
                                                  N_TOK, F_DIM, H_DIM);
  }

  // 4) out = g @ w2   [N_TOK, H] f32
  {
    dim3 grid(N_TOK / 64, H_DIM / 128);   // x: M (fastest), y: N
    gemm_bf16_kernel<1><<<grid, 256, 0, stream>>>(g, w2t, nullptr, nullptr, out,
                                                  N_TOK, H_DIM, F_DIM);
  }

  // 5) add selected-expert LoRA contributions
  lora_kernel<<<N_TOK / 8, 256, 0, stream>>>(x, lA, lB, topw, topi, out);
}
